// MultiHeadedAttention_55765855371429
// MI455X (gfx1250) — matrix operand order repaired and checked by reading, not compiled, not run on hardware
//
#include <hip/hip_runtime.h>
#include <hip/hip_bf16.h>
#include <stdint.h>

// Problem constants (match reference)
constexpr int cB   = 2;
constexpr int cS   = 2048;
constexpr int cHID = 4096;
constexpr int cNH  = 32;
constexpr int cNKV = 8;
constexpr int cD   = 128;

typedef __attribute__((ext_vector_type(16))) __bf16 v16bf;
typedef __attribute__((ext_vector_type(8)))  float  v8f;

union Frag16 { v16bf v; uint32_t u[8]; };
union FragC  { v8f   v; float    f[8]; };

__device__ __forceinline__ uint32_t pack_bf16(float a, float b) {
    union { __bf16 h[2]; uint32_t u; } p;
    p.h[0] = (__bf16)a; p.h[1] = (__bf16)b;
    return p.u;   // v_cvt_pk_bf16_f32
}

// ---------------------------------------------------------------------------
// Async global->LDS 16B copy (CDNA5 GLOBAL_LOAD_ASYNC_TO_LDS_B128, ASYNCcnt).
// Falls back to a synchronous copy if the builtin is unavailable.
// ---------------------------------------------------------------------------
#if defined(__AMDGCN__) && __has_builtin(__builtin_amdgcn_global_load_async_to_lds_b128)
#define HAVE_ASYNC_LDS 1
#else
#define HAVE_ASYNC_LDS 0
#endif

typedef int gvec4i __attribute__((vector_size(16)));

__device__ __forceinline__ void async_copy16(const void* g, void* l) {
#if HAVE_ASYNC_LDS
    __builtin_amdgcn_global_load_async_to_lds_b128(
        (__attribute__((address_space(1))) gvec4i*)(uintptr_t)g,
        (__attribute__((address_space(3))) gvec4i*)(uintptr_t)l, 0, 0);
#else
    *reinterpret_cast<uint4*>(l) = *reinterpret_cast<const uint4*>(g);
#endif
}

__device__ __forceinline__ void async_wait() {
#if HAVE_ASYNC_LDS
#if __has_builtin(__builtin_amdgcn_s_wait_asynccnt)
    __builtin_amdgcn_s_wait_asynccnt(0);
#else
    asm volatile("s_wait_asynccnt 0x0" ::: "memory");
#endif
#endif
}

// ---------------------------------------------------------------------------
// One-shot fp32 -> bf16 elementwise convert (4 elems/thread).
// ---------------------------------------------------------------------------
__global__ void convert_bf16_kernel(const float* __restrict__ in,
                                    __bf16* __restrict__ out, size_t n4)
{
    size_t i = (size_t)blockIdx.x * blockDim.x + threadIdx.x;
    if (i >= n4) return;
    float4 f = reinterpret_cast<const float4*>(in)[i];
    uint32_t* o = reinterpret_cast<uint32_t*>(out) + 2 * i;
    o[0] = pack_bf16(f.x, f.y);
    o[1] = pack_bf16(f.z, f.w);
}

// ---------------------------------------------------------------------------
// One-shot transpose-convert: W fp32 [K,N] -> Wt bf16 [N,K]. 32x32 LDS tile.
// ---------------------------------------------------------------------------
__global__ __launch_bounds__(256) void transpose_convert_kernel(
    const float* __restrict__ W, __bf16* __restrict__ Wt, int K, int N)
{
    __shared__ float tile[32][33];
    const int tid = threadIdx.x;
    const int n0 = blockIdx.x * 32;
    const int k0 = blockIdx.y * 32;
#pragma unroll
    for (int r = 0; r < 4; ++r) {
        int idx = tid + r * 256;
        int kl = idx >> 5, nl = idx & 31;
        tile[kl][nl] = W[(size_t)(k0 + kl) * N + n0 + nl];
    }
    __syncthreads();
#pragma unroll
    for (int r = 0; r < 2; ++r) {
        int w  = tid + r * 256;
        int nl = w >> 4;
        int kp = (w & 15) * 2;
        uint32_t pk = pack_bf16(tile[kp][nl], tile[kp + 1][nl]);
        *reinterpret_cast<uint32_t*>(&Wt[(size_t)(n0 + nl) * K + k0 + kp]) = pk;
    }
}

// ---------------------------------------------------------------------------
// One-shot bf16 transpose of V: (B,S,NKV,D) -> [b][hkv][d][s]. 32x32 tile.
// ---------------------------------------------------------------------------
__global__ __launch_bounds__(256) void transpose_v_kernel(
    const __bf16* __restrict__ V, __bf16* __restrict__ Vt)
{
    __shared__ __bf16 tile[32][34];
    const int tid = threadIdx.x;
    const int s0 = blockIdx.x * 32;
    const int d0 = blockIdx.y * 32;
    const int b   = blockIdx.z / cNKV;
    const int hkv = blockIdx.z % cNKV;
#pragma unroll
    for (int r = 0; r < 4; ++r) {
        int idx = tid + r * 256;
        int sl = idx >> 5, dl = idx & 31;
        tile[sl][dl] = V[((size_t)(b * cS + s0 + sl) * cNKV + hkv) * cD + d0 + dl];
    }
    __syncthreads();
#pragma unroll
    for (int r = 0; r < 2; ++r) {
        int w  = tid + r * 256;
        int dl = w >> 4;
        int sp = (w & 15) * 2;
        union { __bf16 h[2]; uint32_t u; } pp;
        pp.h[0] = tile[sp][dl];
        pp.h[1] = tile[sp + 1][dl];
        *reinterpret_cast<uint32_t*>(
            &Vt[((size_t)(b * cNKV + hkv) * cD + d0 + dl) * cS + s0 + sp]) = pp.u;
    }
}

// ---------------------------------------------------------------------------
// Interleaved RoPE on bf16 (B,S,H,D), one dword (pair) per thread.
// ---------------------------------------------------------------------------
__global__ void rope_bf16_kernel(__bf16* __restrict__ q,
                                 const float* __restrict__ cosT,
                                 const float* __restrict__ sinT,
                                 int H, size_t npairs)
{
    size_t idx = (size_t)blockIdx.x * blockDim.x + threadIdx.x;
    if (idx >= npairs) return;
    const int halfD = cD / 2;
    int i = (int)(idx % halfD);
    size_t rem = idx / halfD;          // (b*S + s)*H + h
    int s = (int)((rem / H) % cS);
    float c  = cosT[s * halfD + i];
    float sn = sinT[s * halfD + i];
    uint32_t* qw = reinterpret_cast<uint32_t*>(q);
    union { __bf16 h[2]; uint32_t u; } pp;
    pp.u = qw[idx];
    float x1 = (float)pp.h[0], x2 = (float)pp.h[1];
    qw[idx] = pack_bf16(x1 * c - x2 * sn, x1 * sn + x2 * c);
}

// ---------------------------------------------------------------------------
// GEMM: C[M,N] = A[M,K] * Bt[N,K]^T, all bf16 operands (k-major both sides),
// fp32 WMMA accumulate, OutT output. 256 thr = 8 waves, BM=BN=128, BK=32,
// double-buffered LDS fed by async global->LDS copies.
// ---------------------------------------------------------------------------
#define GBM 128
#define GBN 128
#define GBK 32

template <typename OutT>
__global__ __launch_bounds__(256) void gemm_bf16_wmma(
    const __bf16* __restrict__ A, const __bf16* __restrict__ Bt,
    OutT* __restrict__ C, int M, int N, int K)
{
    __shared__ __bf16 sA[2][GBM * GBK];   // [m][k]
    __shared__ __bf16 sB[2][GBN * GBK];   // [n][k]

    const int tid  = threadIdx.x;
    const int lane = tid & 31;
    const int wave = tid >> 5;
    const int wm   = wave & 3;      // 32 rows each
    const int wn   = wave >> 2;     // 64 cols each
    const int half = lane >> 4;
    const int l16  = lane & 15;

    const int m0 = blockIdx.y * GBM;
    const int n0 = blockIdx.x * GBN;

    FragC acc[2][4];
#pragma unroll
    for (int mt = 0; mt < 2; ++mt)
#pragma unroll
        for (int t = 0; t < 4; ++t)
#pragma unroll
            for (int j = 0; j < 8; ++j) acc[mt][t].f[j] = 0.0f;

    // 128 rows x 32 k bf16 = 512 x 16B per tile -> 2 copies/thread each side
    const int mrow = tid >> 2;
    const int kq8  = (tid & 3) * 8;

    auto stage_async = [&](int k0, int buf) {
#pragma unroll
        for (int r = 0; r < 2; ++r) {
            int m = mrow + r * 64;
            async_copy16(&A [(size_t)(m0 + m) * K + k0 + kq8], &sA[buf][m * GBK + kq8]);
            async_copy16(&Bt[(size_t)(n0 + m) * K + k0 + kq8], &sB[buf][m * GBK + kq8]);
        }
    };

    auto compute = [&](int buf) {
        Frag16 af[2];
#pragma unroll
        for (int mt = 0; mt < 2; ++mt) {
            int m = wm * 32 + mt * 16 + l16;
#pragma unroll
            for (int v2 = 0; v2 < 8; ++v2) {
                int kb = (v2 < 4 ? 2 * v2 : 16 + 2 * (v2 - 4)) + half * 8;
                af[mt].u[v2] = *reinterpret_cast<const uint32_t*>(&sA[buf][m * GBK + kb]);
            }
        }
#pragma unroll
        for (int t = 0; t < 4; ++t) {
            Frag16 bfm;
            int n = wn * 64 + t * 16 + l16;
#pragma unroll
            for (int v2 = 0; v2 < 8; ++v2) {
                int kb = (v2 < 4 ? 2 * v2 : 16 + 2 * (v2 - 4)) + half * 8;
                bfm.u[v2] = *reinterpret_cast<const uint32_t*>(&sB[buf][n * GBK + kb]);
            }
#pragma unroll
            for (int mt = 0; mt < 2; ++mt)
                acc[mt][t].v = __builtin_amdgcn_wmma_f32_16x16x32_bf16(
                    false, af[mt].v, false, bfm.v, (short)0, acc[mt][t].v, false, false);
        }
    };

    // Pipeline: async-stage tile t+1 into the inactive buffer while WMMAs
    // consume tile t from the active one.
    stage_async(0, 0);
    async_wait();
    __syncthreads();
    const int nkt = K / GBK;
    for (int t = 0; t < nkt; ++t) {
        int cur = t & 1;
        if (t + 1 < nkt) stage_async((t + 1) * GBK, cur ^ 1);
        compute(cur);
        if (t + 1 < nkt) async_wait();
        __syncthreads();
    }

#pragma unroll
    for (int mt = 0; mt < 2; ++mt)
#pragma unroll
        for (int t = 0; t < 4; ++t)
#pragma unroll
            for (int j = 0; j < 8; ++j) {
                int m = m0 + wm * 32 + mt * 16 + j + half * 8;
                int n = n0 + wn * 64 + t * 16 + l16;
                C[(size_t)m * N + n] = (OutT)acc[mt][t].f[j];
            }
}

// ---------------------------------------------------------------------------
// Flash attention (causal, GQA), all-bf16 streams. Block = 8 waves = 128
// queries of one (b,h). K tile and pre-transposed V tile stage via async
// global->LDS b128 copies.
// ---------------------------------------------------------------------------
__global__ __launch_bounds__(256) void flash_attn_wmma(
    const __bf16* __restrict__ Q, const __bf16* __restrict__ Kg,
    const __bf16* __restrict__ Vt, __bf16* __restrict__ O)
{
    __shared__ __bf16 sK[32 * 128];       // [key][d]
    __shared__ __bf16 sV[128 * 32];       // [d][key]
    __shared__ __bf16 sP[8 * 16 * 32];    // per-wave P tile [m][k32]

    const int tid  = threadIdx.x;
    const int lane = tid & 31;
    const int wave = tid >> 5;            // 0..7
    const int half = lane >> 4;
    const int l16  = lane & 15;

    const int h   = blockIdx.y;
    const int b   = blockIdx.z;
    const int hkv = h / (cNH / cNKV);
    const int q0  = blockIdx.x * 128;
    const int qw  = q0 + wave * 16;

    // Q fragments: direct dword gathers (pairs already adjacent bf16)
    Frag16 qf[4];
    {
        const int m = qw + l16;
        const __bf16* qrow = Q + ((size_t)(b * cS + m) * cNH + h) * cD;
#pragma unroll
        for (int kk = 0; kk < 4; ++kk)
#pragma unroll
            for (int v2 = 0; v2 < 8; ++v2) {
                int kb = kk * 32 + (v2 < 4 ? 2 * v2 : 16 + 2 * (v2 - 4)) + half * 8;
                qf[kk].u[v2] = *reinterpret_cast<const uint32_t*>(&qrow[kb]);
            }
    }

    FragC acc[8];
#pragma unroll
    for (int t = 0; t < 8; ++t)
#pragma unroll
        for (int j = 0; j < 8; ++j) acc[t].f[j] = 0.0f;

    float mrow[8], lrow[8];
#pragma unroll
    for (int j = 0; j < 8; ++j) { mrow[j] = -3.0e38f; lrow[j] = 0.0f; }

    const float scale = 0.08838834764831845f;   // 1/sqrt(128)
    const int nchunks = (q0 + 128) / 32;

    const __bf16* kbaseptr = Kg + ((size_t)b * cS * cNKV + hkv) * cD;
    const __bf16* vbaseptr = Vt + ((size_t)(b * cNKV + hkv) * cD) * cS;

    for (int kc = 0; kc < nchunks; ++kc) {
        const int kbase = kc * 32;
        // Async-stage K chunk [32 key][128 d]: 512 x 16B / 256 thr = 2 each
#pragma unroll
        for (int r = 0; r < 2; ++r) {
            int idx = tid + r * 256;
            int key = idx >> 4;
            int q4  = (idx & 15) * 8;
            async_copy16(&kbaseptr[(size_t)(kbase + key) * cNKV * cD + q4],
                         &sK[key * 128 + q4]);
        }
        // Async-stage V chunk [128 d][32 key] from pre-transposed Vt
#pragma unroll
        for (int r = 0; r < 2; ++r) {
            int idx = tid + r * 256;
            int dl  = idx >> 2;
            int q4  = (idx & 3) * 8;
            async_copy16(&vbaseptr[(size_t)dl * cS + kbase + q4],
                         &sV[dl * 32 + q4]);
        }
        // Prefetch the next chunk into L2 (global_prefetch_b8)
        if (kc + 1 < nchunks) {
            __builtin_prefetch(&kbaseptr[(size_t)(kbase + 32 + (tid >> 3)) * cNKV * cD
                                         + (size_t)(tid & 7) * 16], 0, 0);
            __builtin_prefetch(&vbaseptr[(size_t)(tid >> 1) * cS + kbase + 32
                                         + (size_t)(tid & 1) * 16], 0, 0);
        }
        async_wait();
        __syncthreads();

        // S = Q * K^T  (two 16-key tiles, d chained 4x32)
        FragC s0, s1;
#pragma unroll
        for (int j = 0; j < 8; ++j) { s0.f[j] = 0.0f; s1.f[j] = 0.0f; }
#pragma unroll
        for (int kk = 0; kk < 4; ++kk) {
            Frag16 bf0, bf1;
#pragma unroll
            for (int v2 = 0; v2 < 8; ++v2) {
                int d = kk * 32 + (v2 < 4 ? 2 * v2 : 16 + 2 * (v2 - 4)) + half * 8;
                bf0.u[v2] = *reinterpret_cast<const uint32_t*>(&sK[(l16)      * 128 + d]);
                bf1.u[v2] = *reinterpret_cast<const uint32_t*>(&sK[(16 + l16) * 128 + d]);
            }
            s0.v = __builtin_amdgcn_wmma_f32_16x16x32_bf16(
                false, qf[kk].v, false, bf0.v, (short)0, s0.v, false, false);
            s1.v = __builtin_amdgcn_wmma_f32_16x16x32_bf16(
                false, qf[kk].v, false, bf1.v, (short)0, s1.v, false, false);
        }

        // Causal mask + scale + online softmax (C layout: row=j+8*half, col=l16)
        float mnew[8];
#pragma unroll
        for (int j = 0; j < 8; ++j) {
            int qg = qw + j + half * 8;
            float a0 = s0.f[j] * scale;
            float a1 = s1.f[j] * scale;
            if (kbase + l16      > qg) a0 = -1.0e9f;
            if (kbase + 16 + l16 > qg) a1 = -1.0e9f;
            s0.f[j] = a0; s1.f[j] = a1;
            float r = fmaxf(a0, a1);
#pragma unroll
            for (int off = 1; off < 16; off <<= 1)
                r = fmaxf(r, __shfl_xor(r, off, 32));
            mnew[j] = fmaxf(mrow[j], r);
        }
#pragma unroll
        for (int j = 0; j < 8; ++j) {
            float al = __expf(mrow[j] - mnew[j]);
            float p0 = __expf(s0.f[j] - mnew[j]);
            float p1 = __expf(s1.f[j] - mnew[j]);
            float rs = p0 + p1;
#pragma unroll
            for (int off = 1; off < 16; off <<= 1)
                rs += __shfl_xor(rs, off, 32);
            lrow[j] = lrow[j] * al + rs;
            mrow[j] = mnew[j];
            int row = j + half * 8;
            sP[wave * 512 + row * 32 + l16]      = (__bf16)p0;
            sP[wave * 512 + row * 32 + 16 + l16] = (__bf16)p1;
#pragma unroll
            for (int dt = 0; dt < 8; ++dt) acc[dt].f[j] *= al;
        }

        // Re-gather P as an A-fragment (16x32)
        Frag16 pf;
#pragma unroll
        for (int v2 = 0; v2 < 8; ++v2) {
            int kb = (v2 < 4 ? 2 * v2 : 16 + 2 * (v2 - 4)) + half * 8;
            pf.u[v2] = *reinterpret_cast<const uint32_t*>(&sP[wave * 512 + l16 * 32 + kb]);
        }

        // ctx += P * V
#pragma unroll
        for (int dt = 0; dt < 8; ++dt) {
            Frag16 vf;
#pragma unroll
            for (int v2 = 0; v2 < 8; ++v2) {
                int kk2 = (v2 < 4 ? 2 * v2 : 16 + 2 * (v2 - 4)) + half * 8;
                vf.u[v2] = *reinterpret_cast<const uint32_t*>(&sV[(dt * 16 + l16) * 32 + kk2]);
            }
            acc[dt].v = __builtin_amdgcn_wmma_f32_16x16x32_bf16(
                false, pf.v, false, vf.v, (short)0, acc[dt].v, false, false);
        }
        __syncthreads();
    }

    // Normalize and write ctx bf16 (B,S,NH,D)
#pragma unroll
    for (int j = 0; j < 8; ++j) {
        float inv = 1.0f / lrow[j];
        int qg = qw + j + half * 8;
        __bf16* orow = O + ((size_t)(b * cS + qg) * cNH + h) * cD;
#pragma unroll
        for (int dt = 0; dt < 8; ++dt)
            orow[dt * 16 + l16] = (__bf16)(acc[dt].f[j] * inv);
    }
}

// ---------------------------------------------------------------------------
// Launch
// ---------------------------------------------------------------------------
extern "C" void kernel_launch(void* const* d_in, const int* in_sizes, int n_in,
                              void* d_out, int out_size, void* d_ws, size_t ws_size,
                              hipStream_t stream)
{
    const float* x    = (const float*)d_in[0];
    const float* Wq   = (const float*)d_in[1];
    const float* Wk   = (const float*)d_in[2];
    const float* Wv   = (const float*)d_in[3];
    const float* Wo   = (const float*)d_in[4];
    const float* cosT = (const float*)d_in[5];
    const float* sinT = (const float*)d_in[6];
    float* out = (float*)d_out;

    const size_t xE  = (size_t)cB * cS * cHID;          // 16M
    const size_t qE  = (size_t)cB * cS * cNH  * cD;     // 16M
    const size_t kvE = (size_t)cB * cS * cNKV * cD;     // 4M
    const size_t wqE = (size_t)cHID * cNH  * cD;        // 16M
    const size_t wkE = (size_t)cHID * cNKV * cD;        // 4M

    __bf16* p = (__bf16*)d_ws;
    __bf16* xb   = p; p += xE;
    __bf16* Wqt  = p; p += wqE;
    __bf16* Wkt  = p; p += wkE;
    __bf16* Wvt  = p; p += wkE;
    __bf16* Wot  = p; p += wqE;
    __bf16* qb   = p; p += qE;
    __bf16* kb   = p; p += kvE;
    __bf16* vb   = p; p += kvE;
    __bf16* vt   = p; p += kvE;
    __bf16* ctxb = p; p += qE;

    const int M = cB * cS;   // 4096

    // One-shot conversions / layout changes
    convert_bf16_kernel<<<(unsigned)((xE / 4 + 255) / 256), 256, 0, stream>>>(x, xb, xE / 4);
    transpose_convert_kernel<<<dim3(cNH * cD / 32,  cHID / 32), 256, 0, stream>>>(Wq, Wqt, cHID, cNH * cD);
    transpose_convert_kernel<<<dim3(cNKV * cD / 32, cHID / 32), 256, 0, stream>>>(Wk, Wkt, cHID, cNKV * cD);
    transpose_convert_kernel<<<dim3(cNKV * cD / 32, cHID / 32), 256, 0, stream>>>(Wv, Wvt, cHID, cNKV * cD);
    transpose_convert_kernel<<<dim3(cHID / 32, cNH * cD / 32),  256, 0, stream>>>(Wo, Wot, cNH * cD, cHID);

    // Projections (bf16 in, bf16 out)
    gemm_bf16_wmma<__bf16><<<dim3((cNH * cD) / GBN,  M / GBM), 256, 0, stream>>>(xb, Wqt, qb, M, cNH * cD,  cHID);
    gemm_bf16_wmma<__bf16><<<dim3((cNKV * cD) / GBN, M / GBM), 256, 0, stream>>>(xb, Wkt, kb, M, cNKV * cD, cHID);
    gemm_bf16_wmma<__bf16><<<dim3((cNKV * cD) / GBN, M / GBM), 256, 0, stream>>>(xb, Wvt, vb, M, cNKV * cD, cHID);

    // RoPE on q and k (bf16 dword pairs)
    size_t nq = (size_t)cB * cS * cNH  * (cD / 2);
    size_t nk = (size_t)cB * cS * cNKV * (cD / 2);
    rope_bf16_kernel<<<(unsigned)((nq + 255) / 256), 256, 0, stream>>>(qb, cosT, sinT, cNH,  nq);
    rope_bf16_kernel<<<(unsigned)((nk + 255) / 256), 256, 0, stream>>>(kb, cosT, sinT, cNKV, nk);

    // V -> [b][hkv][d][s] for straight-copy flash staging
    transpose_v_kernel<<<dim3(cS / 32, cD / 32, cB * cNKV), 256, 0, stream>>>(vb, vt);

    // Causal GQA flash attention (bf16 streams, 128 queries/block)
    flash_attn_wmma<<<dim3(cS / 128, cNH, cB), 256, 0, stream>>>(qb, kb, vt, ctxb);

    // Output projection (bf16 in, fp32 out)
    gemm_bf16_wmma<float><<<dim3(cHID / GBN, M / GBM), 256, 0, stream>>>(ctxb, Wot, out, M, cHID, cHID);

    (void)in_sizes; (void)n_in; (void)out_size; (void)ws_size;
}
